// Attention_20512763805926
// MI455X (gfx1250) — matrix operand order repaired and checked by reading, not compiled, not run on hardware
//
#include <hip/hip_runtime.h>
#include <hip/hip_bf16.h>

// ---------------------------------------------------------------------------
// MI455X (gfx1250, wave32) 3D-grid attention:
//   qkv = x @ Wqkv + b ; flash-attention per head ; out = O @ Wproj + b
// All GEMMs + attention einsums via v_wmma_f32_16x16x32_f16 (f16 in, f32 acc).
// GEMMs: 64x64 register-blocked tile / wave (16 WMMA per 16 b128 loads).
// Attention: 4 waves x 32 queries per block; K/V tiles staged in LDS once per
// block (4x L2 traffic reduction), fragments fed from the DS pipe, next tile
// prefetched with global_prefetch_b8.
// ---------------------------------------------------------------------------

typedef __attribute__((ext_vector_type(16))) _Float16 v16h;
typedef __attribute__((ext_vector_type(8)))  _Float16 v8h;
typedef __attribute__((ext_vector_type(8)))  float    v8f;

#define NTOK 4096
#define CDIM 768
#define HD   64
#define NH   12
#define C3   (3*CDIM)

__device__ inline v8f wmma16(v16h a, v16h b, v8f c) {
  return __builtin_amdgcn_wmma_f32_16x16x32_f16(false, a, false, b,
                                                (short)0, c, false, false);
}

// 16 contiguous halfs (two b128 loads).
__device__ inline v16h load16(const _Float16* p) {
  v8h a = *(const v8h*)p;
  v8h b = *(const v8h*)(p + 8);
  return __builtin_shufflevector(a, b, 0,1,2,3,4,5,6,7,8,9,10,11,12,13,14,15);
}

// A-matrix fragment (16x32 f16): lane holds row M = lane%16.
// lane<16: K = kk+0..7 , kk+16..23 ; lane>=16: kk+8..15 , kk+24..31.
__device__ inline v16h loadA(const _Float16* rowbase, int kk, int lh) {
  const _Float16* p = rowbase + kk + lh * 8;
  v8h a = *(const v8h*)p;
  v8h b = *(const v8h*)(p + 16);
  return __builtin_shufflevector(a, b, 0,1,2,3,4,5,6,7,8,9,10,11,12,13,14,15);
}

union U16 { int u[8]; v16h v; };
union H8  { _Float16 h[8]; v8h v; };
union PK  { _Float16 h[2]; int u; };

__device__ inline int pk2(float a, float b) {
  PK x; x.h[0] = (_Float16)a; x.h[1] = (_Float16)b; return x.u;
}

__device__ inline v8f vzero() {
  v8f z;
  #pragma unroll
  for (int i = 0; i < 8; ++i) z[i] = 0.f;
  return z;
}

// --------------------------- conversion pre-pass ---------------------------
__global__ void cvt_f16(const float* __restrict__ s, _Float16* __restrict__ d, int n) {
  int i = blockIdx.x * blockDim.x + threadIdx.x;
  if (i < n) d[i] = (_Float16)s[i];
}
// dst[n][k] = (f16) src[k][n]   (src is K x N row-major)
__global__ void cvt_tr(const float* __restrict__ s, _Float16* __restrict__ d, int K, int N) {
  int i = blockIdx.x * blockDim.x + threadIdx.x;
  if (i < K * N) {
    int k = i / N, n = i % N;
    d[(size_t)n * K + k] = (_Float16)s[i];
  }
}

// 64x64 register-blocked WMMA GEMM core.
__device__ inline void gemm64(const _Float16* a0, const _Float16* b0,
                              v8f c[4][4], int ln, int lh) {
  const _Float16* ar[4]; const _Float16* br[4];
  #pragma unroll
  for (int s = 0; s < 4; ++s) {
    ar[s] = a0 + (size_t)(s * 16) * CDIM;
    br[s] = b0 + (size_t)(s * 16) * CDIM;
  }
  #pragma unroll 2
  for (int kk = 0; kk < CDIM; kk += 32) {
    v16h a[4], b[4];
    #pragma unroll
    for (int s = 0; s < 4; ++s) a[s] = loadA(ar[s], kk, lh);
    #pragma unroll
    for (int s = 0; s < 4; ++s) b[s] = loadA(br[s], kk, lh);
    #pragma unroll
    for (int sm = 0; sm < 4; ++sm)
      #pragma unroll
      for (int sn = 0; sn < 4; ++sn)
        c[sm][sn] = wmma16(a[sm], b[sn], c[sm][sn]);
  }
}

// ------------------------------- QKV GEMM ----------------------------------
__global__ __launch_bounds__(32)
void qkv_kernel(const _Float16* __restrict__ xh, const _Float16* __restrict__ wqT,
                const float* __restrict__ bq,
                _Float16* __restrict__ Qh, _Float16* __restrict__ Kh,
                _Float16* __restrict__ Vt) {
  const int lane = threadIdx.x, ln = lane & 15, lh = lane >> 4;
  const int tm = blockIdx.x;            // 64-token tile   (64)
  const int tn = blockIdx.y;            // 64-column tile  (36)

  v8f c[4][4];
  #pragma unroll
  for (int i = 0; i < 4; ++i)
    #pragma unroll
    for (int j = 0; j < 4; ++j) c[i][j] = vzero();

  gemm64(xh  + (size_t)(tm * 64 + ln) * CDIM,
         wqT + (size_t)(tn * 64 + ln) * CDIM, c, ln, lh);

  const int colbase = tn * 64;                 // multiple of 64
  const int which   = colbase / CDIM;          // wave-uniform: 0=Q 1=K 2=V
  const int head    = (colbase % CDIM) / HD;   // wave-uniform

  #pragma unroll
  for (int sn = 0; sn < 4; ++sn) {
    const int dim  = sn * 16 + ln;             // 0..63 within head
    const float bias = bq[colbase + dim];
    #pragma unroll
    for (int sm = 0; sm < 4; ++sm) {
      const int tok0 = tm * 64 + sm * 16 + 8 * lh;   // 8 consecutive tokens
      if (which == 2) {
        H8 o;
        #pragma unroll
        for (int r = 0; r < 8; ++r) o.h[r] = (_Float16)(c[sm][sn][r] + bias);
        *(v8h*)(Vt + ((size_t)(head * HD + dim)) * NTOK + tok0) = o.v;
      } else if (which == 0) {
        #pragma unroll
        for (int r = 0; r < 8; ++r)
          Qh[((size_t)(head * NTOK + tok0 + r)) * HD + dim] =
              (_Float16)((c[sm][sn][r] + bias) * 0.125f);
      } else {
        #pragma unroll
        for (int r = 0; r < 8; ++r)
          Kh[((size_t)(head * NTOK + tok0 + r)) * HD + dim] =
              (_Float16)(c[sm][sn][r] + bias);
      }
    }
  }
}

// --------------------------- flash attention -------------------------------
// Per 32-key block for one 16-query tile: per-lane online softmax and
// O^T += V^T . P  with P built as B-operand via packed half-wave shuffles.
__device__ inline void softmax_pv(v8f s0, v8f s1, float& m_run, float& l_run,
                                  v8f acc[4], const v16h av[4], int lh) {
  float mx = s0[0];
  #pragma unroll
  for (int i = 1; i < 8; ++i) mx = fmaxf(mx, s0[i]);
  #pragma unroll
  for (int i = 0; i < 8; ++i) mx = fmaxf(mx, s1[i]);
  mx = fmaxf(mx, __shfl_xor(mx, 16, 32));
  const float m_new = fmaxf(m_run, mx);
  const float corr  = __expf(m_run - m_new);

  float e0[8], e1[8], ssum = 0.f;
  #pragma unroll
  for (int i = 0; i < 8; ++i) { e0[i] = __expf(s0[i] - m_new); ssum += e0[i]; }
  #pragma unroll
  for (int i = 0; i < 8; ++i) { e1[i] = __expf(s1[i] - m_new); ssum += e1[i]; }
  ssum += __shfl_xor(ssum, 16, 32);
  l_run = l_run * corr + ssum;
  m_run = m_new;
  #pragma unroll
  for (int t = 0; t < 4; ++t)
    #pragma unroll
    for (int i = 0; i < 8; ++i) acc[t][i] *= corr;

  int p0[4], p1[4], q0[4], q1[4];
  #pragma unroll
  for (int i = 0; i < 4; ++i) { p0[i] = pk2(e0[2*i], e0[2*i+1]);
                                p1[i] = pk2(e1[2*i], e1[2*i+1]); }
  #pragma unroll
  for (int i = 0; i < 4; ++i) { q0[i] = __shfl_xor(p0[i], 16, 32);
                                q1[i] = __shfl_xor(p1[i], 16, 32); }
  U16 bp;
  #pragma unroll
  for (int i = 0; i < 4; ++i) {
    bp.u[i]     = (lh == 0) ? p0[i] : q1[i];   // keys 0..7  / 16..23
    bp.u[i + 4] = (lh == 0) ? q0[i] : p1[i];   // keys 8..15 / 24..31
  }
  #pragma unroll
  for (int t = 0; t < 4; ++t) acc[t] = wmma16(av[t], bp.v, acc[t]);
}

// 4 waves x 32 queries per block; K/V 32-key tiles staged in LDS per block.
__global__ __launch_bounds__(128)
void attn_kernel(const _Float16* __restrict__ Qh, const _Float16* __restrict__ Kh,
                 const _Float16* __restrict__ Vt, _Float16* __restrict__ Oo) {
  __shared__ _Float16 lk[32 * HD];    // K tile:  [tok 0..31][dim 0..63]
  __shared__ _Float16 lv[HD * 32];    // V^T tile:[dim 0..63][tok 0..31]

  const int tid  = threadIdx.x;       // 0..127
  const int wid  = tid >> 5;          // wave id 0..3
  const int lane = tid & 31, ln = lane & 15, lh = lane >> 4;
  const int h  = blockIdx.y;          // 0..11
  const int q0 = blockIdx.x * 128 + wid * 32;   // this wave's 32 queries

  // Q B-fragments (constant over key loop)
  const _Float16* qa = Qh + ((size_t)(h * NTOK + q0 + ln)) * HD;
  const _Float16* qb = qa + (size_t)16 * HD;
  const v16h bq0a = loadA(qa,  0, lh);
  const v16h bq1a = loadA(qa, 32, lh);
  const v16h bq0b = loadA(qb,  0, lh);
  const v16h bq1b = loadA(qb, 32, lh);

  // Per-thread cooperative staging chunk (16 halfs of each tile)
  const int fk = tid * 16;                                  // 0..2047
  const _Float16* gk0 = Kh + (size_t)h * NTOK * HD + (size_t)(fk >> 6) * HD + (fk & 63);
  const _Float16* gv0 = Vt + (size_t)h * HD * NTOK + (size_t)(fk >> 5) * NTOK + (fk & 31);

  float ma = -1e30f, la = 0.f, mb = -1e30f, lb = 0.f;
  v8f accA[4], accB[4];
  #pragma unroll
  for (int t = 0; t < 4; ++t) { accA[t] = vzero(); accB[t] = vzero(); }

  for (int kb = 0; kb < NTOK; kb += 32) {
    // ---- stage K (32x64) and V^T (64x32) tiles into LDS ----
    const _Float16* gk = gk0 + (size_t)kb * HD;
    const _Float16* gv = gv0 + kb;
    v8h r0 = *(const v8h*)gk, r1 = *(const v8h*)(gk + 8);
    v8h r2 = *(const v8h*)gv, r3 = *(const v8h*)(gv + 8);
    *(v8h*)(lk + fk)     = r0;  *(v8h*)(lk + fk + 8) = r1;
    *(v8h*)(lv + fk)     = r2;  *(v8h*)(lv + fk + 8) = r3;
    __syncthreads();                       // tiles visible to all 4 waves

    // warm the caches for the next key block while we compute
    if (kb + 32 < NTOK) {
      __builtin_prefetch(gk + (size_t)32 * HD, 0, 0);
      __builtin_prefetch(gv + 32, 0, 0);
    }

    // ---- fragments from LDS ----
    const _Float16* kr0 = lk + (size_t)ln * HD;
    const _Float16* kr1 = lk + (size_t)(16 + ln) * HD;
    const v16h ka0 = loadA(kr0,  0, lh);
    const v16h ka1 = loadA(kr0, 32, lh);
    const v16h kc0 = loadA(kr1,  0, lh);
    const v16h kc1 = loadA(kr1, 32, lh);
    v16h av[4];
    #pragma unroll
    for (int t = 0; t < 4; ++t)
      av[t] = load16(lv + (size_t)(t * 16 + ln) * 32 + lh * 16);

    // ---- S^T tiles (keys x queries), fp32 accum ----
    v8f s0a = vzero(), s1a = vzero(), s0b = vzero(), s1b = vzero();
    s0a = wmma16(ka0, bq0a, s0a); s0a = wmma16(ka1, bq1a, s0a);
    s1a = wmma16(kc0, bq0a, s1a); s1a = wmma16(kc1, bq1a, s1a);
    s0b = wmma16(ka0, bq0b, s0b); s0b = wmma16(ka1, bq1b, s0b);
    s1b = wmma16(kc0, bq0b, s1b); s1b = wmma16(kc1, bq1b, s1b);

    softmax_pv(s0a, s1a, ma, la, accA, av, lh);
    softmax_pv(s0b, s1b, mb, lb, accB, av, lh);

    __syncthreads();                       // done reading tiles
  }

  const float ia = 1.f / la, ib = 1.f / lb;
  _Float16* oa = Oo + (size_t)(q0 + ln) * CDIM + h * HD + lh * 8;
  _Float16* ob = Oo + (size_t)(q0 + 16 + ln) * CDIM + h * HD + lh * 8;
  #pragma unroll
  for (int t = 0; t < 4; ++t) {
    H8 o;
    #pragma unroll
    for (int i = 0; i < 8; ++i) o.h[i] = (_Float16)(accA[t][i] * ia);
    *(v8h*)(oa + t * 16) = o.v;
    #pragma unroll
    for (int i = 0; i < 8; ++i) o.h[i] = (_Float16)(accB[t][i] * ib);
    *(v8h*)(ob + t * 16) = o.v;
  }
}

// ------------------------------ output proj --------------------------------
__global__ __launch_bounds__(32)
void proj_kernel(const _Float16* __restrict__ Oo, const _Float16* __restrict__ wpT,
                 const float* __restrict__ bp, float* __restrict__ out) {
  const int lane = threadIdx.x, ln = lane & 15, lh = lane >> 4;
  const int tm = blockIdx.x;            // 64-token tile (64)
  const int tn = blockIdx.y;            // 64-col tile   (12)

  v8f c[4][4];
  #pragma unroll
  for (int i = 0; i < 4; ++i)
    #pragma unroll
    for (int j = 0; j < 4; ++j) c[i][j] = vzero();

  gemm64(Oo  + (size_t)(tm * 64 + ln) * CDIM,
         wpT + (size_t)(tn * 64 + ln) * CDIM, c, ln, lh);

  #pragma unroll
  for (int sn = 0; sn < 4; ++sn) {
    const int col = tn * 64 + sn * 16 + ln;
    const float bias = bp[col];
    #pragma unroll
    for (int sm = 0; sm < 4; ++sm) {
      const int tok0 = tm * 64 + sm * 16 + 8 * lh;
      #pragma unroll
      for (int r = 0; r < 8; ++r)
        out[(size_t)(tok0 + r) * CDIM + col] = c[sm][sn][r] + bias;
    }
  }
}

// ------------------------------- launcher ----------------------------------
extern "C" void kernel_launch(void* const* d_in, const int* in_sizes, int n_in,
                              void* d_out, int out_size, void* d_ws, size_t ws_size,
                              hipStream_t stream) {
  const float* x      = (const float*)d_in[0];
  const float* w_qkv  = (const float*)d_in[1];
  const float* b_qkv  = (const float*)d_in[2];
  const float* w_proj = (const float*)d_in[3];
  const float* b_proj = (const float*)d_in[4];
  float* out = (float*)d_out;

  _Float16* ws  = (_Float16*)d_ws;
  _Float16* xh  = ws;                                     // 4096*768
  _Float16* wqT = xh  + (size_t)NTOK * CDIM;              // 2304*768
  _Float16* wpT = wqT + (size_t)C3   * CDIM;              // 768*768
  _Float16* Qh  = wpT + (size_t)CDIM * CDIM;              // 12*4096*64
  _Float16* Kh  = Qh  + (size_t)NH * NTOK * HD;
  _Float16* Vt  = Kh  + (size_t)NH * NTOK * HD;
  _Float16* Oo  = Vt  + (size_t)NH * NTOK * HD;           // 4096*768

  const int nx = NTOK * CDIM;
  cvt_f16<<<(nx + 255) / 256, 256, 0, stream>>>(x, xh, nx);
  cvt_tr <<<(CDIM * C3   + 255) / 256, 256, 0, stream>>>(w_qkv,  wqT, CDIM, C3);
  cvt_tr <<<(CDIM * CDIM + 255) / 256, 256, 0, stream>>>(w_proj, wpT, CDIM, CDIM);

  qkv_kernel <<<dim3(NTOK / 64, C3 / 64),   32, 0, stream>>>(xh, wqT, b_qkv, Qh, Kh, Vt);
  attn_kernel<<<dim3(NTOK / 128, NH),      128, 0, stream>>>(Qh, Kh, Vt, Oo);
  proj_kernel<<<dim3(NTOK / 64, CDIM / 64), 32, 0, stream>>>(Oo, wpT, b_proj, out);
}
